// DeepseekV2AttentionMLA_88235808129210
// MI455X (gfx1250) — matrix operand order repaired and checked by reading, not compiled, not run on hardware
//
#include <hip/hip_runtime.h>
#include <hip/hip_bf16.h>

// ---------------------------------------------------------------------------
// DeepSeek-V2 MLA prefill for gfx1250 (MI455X), bf16 WMMA everywhere.
// T=2048, HID=5120, H=16, NOPE=128, ROPE=64, VDIM=128, QLORA=1536, KVLORA=512
// Round 2: double-buffered GEMM staging + TDM (tensor_load_to_lds) staging of
// the attention K tile (async-tensor path, TENSORcnt-tracked).
// ---------------------------------------------------------------------------

typedef __attribute__((ext_vector_type(16))) __bf16 v16bf;
typedef __attribute__((ext_vector_type(8)))  __bf16 v8bf;
typedef __attribute__((ext_vector_type(2)))  __bf16 v2bf;
typedef __attribute__((ext_vector_type(8)))  float  v8f;
typedef __attribute__((ext_vector_type(4)))  unsigned int v4u;
typedef __attribute__((ext_vector_type(8)))  int v8i;
typedef __attribute__((ext_vector_type(4)))  int v4i;

#define T_LEN   2048
#define HID_D   5120
#define NHEAD   16
#define NOPE_D  128
#define ROPE_D  64
#define VDIM_D  128
#define QLORA_D 1536
#define KVLORA_D 512
#define FUSED_N (QLORA_D + KVLORA_D + ROPE_D)   // 2112
#define QB_N    (NHEAD * (NOPE_D + ROPE_D))     // 3072
#define KVB_N   (NHEAD * (NOPE_D + VDIM_D))     // 4096
#define QK_D    (NOPE_D + ROPE_D)               // 192
#define ATTN_SCALE 0.07216878364870322f         // 192^-0.5

#if defined(__has_builtin)
#if __has_builtin(__builtin_amdgcn_tensor_load_to_lds) && \
    __has_builtin(__builtin_amdgcn_s_wait_tensorcnt)
#define USE_TDM 1
#endif
#endif

__device__ __forceinline__ v8f v8f_zero() {
  v8f z;
#pragma unroll
  for (int i = 0; i < 8; ++i) z[i] = 0.f;
  return z;
}

__device__ __forceinline__ v8f wmma_bf16(v16bf a, v16bf b, v8f c) {
  // v_wmma_f32_16x16x32_bf16  D = A*B + C
  return __builtin_amdgcn_wmma_f32_16x16x32_bf16(
      /*neg_a=*/false, a, /*neg_b=*/false, b,
      /*c_mod=*/(short)0, c, /*reuse_a=*/false, /*reuse_b=*/false);
}

// A-operand fragment (16-bit A 16x32 layout): lane half khalf holds
// K = khalf*8 + {0..7} in elems 0..7 and K = 16 + khalf*8 + {0..7} in 8..15.
__device__ __forceinline__ v16bf load_frag_a(const __bf16* row, int khalf) {
  const v8bf lo = *(const v8bf*)(row + khalf * 8);
  const v8bf hi = *(const v8bf*)(row + khalf * 8 + 16);
  v16bf f;
#pragma unroll
  for (int e = 0; e < 8; ++e) { f[e] = lo[e]; f[8 + e] = hi[e]; }
  return f;
}

// B-operand fragment (16-bit B 32x16 layout): lane half khalf holds
// K = khalf*8 + {0..7} in elems 0..7 and K = 16 + khalf*8 + {0..7} in 8..15.
// `row` points at LDS row [N][K].
__device__ __forceinline__ v16bf load_frag_b(const __bf16* row, int khalf) {
  const v8bf lo = *(const v8bf*)(row + khalf * 8);
  const v8bf hi = *(const v8bf*)(row + khalf * 8 + 16);
  v16bf f;
#pragma unroll
  for (int e = 0; e < 8; ++e) { f[e] = lo[e]; f[8 + e] = hi[e]; }
  return f;
}

// ---------------------------------------------------------------------------
// Generic C(MxN,f32) = A(MxK) @ B(KxN,f32) with bf16 WMMA.
// Block tile 128x128, BK=32, 8 waves (4x2), each wave does 2x4 16x16 tiles.
// Double-buffered LDS: stage k+1 while computing k, one barrier/iteration.
// ---------------------------------------------------------------------------
template <typename AT>
__global__ __launch_bounds__(256) void gemm_wb(
    const AT* __restrict__ A, const float* __restrict__ B,
    float* __restrict__ C, int M, int N, int K) {
  __shared__ alignas(16) __bf16 lA[2][128][40];  // [m][k], 80B row stride
  __shared__ alignas(16) __bf16 lB[2][128][40];  // [n][k] (transposed stage)
  const int bm = blockIdx.y * 128, bn = blockIdx.x * 128;
  const int tid = threadIdx.x;
  const int wid = tid >> 5, lane = tid & 31;
  const int l16 = lane & 15, khalf = lane >> 4;
  const int wm = (wid & 3) * 32, wn = (wid >> 2) * 64;

  v8f acc[2][4];
#pragma unroll
  for (int i = 0; i < 2; ++i)
#pragma unroll
    for (int j = 0; j < 4; ++j) acc[i][j] = v8f_zero();

  auto stage = [&](int buf, int k0) {
    // A tile 128x32 (convert to bf16)
    for (int i = tid; i < 128 * 32; i += 256) {
      int r = i >> 5, c = i & 31;
      int gr = bm + r;
      float v = (gr < M) ? (float)A[(size_t)gr * K + (k0 + c)] : 0.f;
      lA[buf][r][c] = (__bf16)v;
    }
    // B tile 32x128, transposed into [n][k]
    for (int i = tid; i < 32 * 128; i += 256) {
      int r = i >> 7, c = i & 127;  // r = k offset, c = n offset
      int gn = bn + c;
      float v = (gn < N) ? B[(size_t)(k0 + r) * N + gn] : 0.f;
      lB[buf][c][r] = (__bf16)v;
    }
  };

  stage(0, 0);
  __syncthreads();

  const int nk = K >> 5;
  for (int kk = 0; kk < nk; ++kk) {
    const int cur = kk & 1;
    if (kk + 1 < nk) stage(cur ^ 1, (kk + 1) << 5);  // overlap with wmma

    v16bf af[2], bfv[4];
#pragma unroll
    for (int mt = 0; mt < 2; ++mt)
      af[mt] = load_frag_a(&lA[cur][wm + mt * 16 + l16][0], khalf);
#pragma unroll
    for (int nt = 0; nt < 4; ++nt)
      bfv[nt] = load_frag_b(&lB[cur][wn + nt * 16 + l16][0], khalf);
#pragma unroll
    for (int mt = 0; mt < 2; ++mt)
#pragma unroll
      for (int nt = 0; nt < 4; ++nt)
        acc[mt][nt] = wmma_bf16(af[mt], bfv[nt], acc[mt][nt]);
    __syncthreads();
  }

  // C/D layout: VGPR r, lanes 0-15 -> row r, lanes 16-31 -> row 8+r; col = l16
#pragma unroll
  for (int mt = 0; mt < 2; ++mt) {
#pragma unroll
    for (int nt = 0; nt < 4; ++nt) {
      int col = bn + wn + nt * 16 + l16;
      if (col >= N) continue;
      int rb = bm + wm + mt * 16 + khalf * 8;
#pragma unroll
      for (int r = 0; r < 8; ++r) {
        int row = rb + r;
        if (row < M) C[(size_t)row * N + col] = acc[mt][nt][r];
      }
    }
  }
}

// ---------------------------------------------------------------------------
// RMSNorm of q_a (1536) and kv_a (512) + RoPE of k_pe (64), one row per block.
// ---------------------------------------------------------------------------
__global__ __launch_bounds__(256) void rms_rope_k(
    const float* __restrict__ qkv, const float* __restrict__ qw,
    const float* __restrict__ kvw, const int* __restrict__ pos,
    __bf16* __restrict__ qan, __bf16* __restrict__ kvan,
    float* __restrict__ kpe) {
  __shared__ float red[8];
  __shared__ float bc[2];
  const int t = blockIdx.x, tid = threadIdx.x;
  const float* row = qkv + (size_t)t * FUSED_N;

  float s = 0.f;
  for (int i = tid; i < QLORA_D; i += 256) { float v = row[i]; s += v * v; }
  s += __shfl_xor(s, 16); s += __shfl_xor(s, 8); s += __shfl_xor(s, 4);
  s += __shfl_xor(s, 2);  s += __shfl_xor(s, 1);
  if ((tid & 31) == 0) red[tid >> 5] = s;
  __syncthreads();
  if (tid == 0) { float a = 0.f; for (int i = 0; i < 8; ++i) a += red[i]; bc[0] = a; }
  __syncthreads();

  float s2 = 0.f;
  for (int i = tid; i < KVLORA_D; i += 256) { float v = row[QLORA_D + i]; s2 += v * v; }
  s2 += __shfl_xor(s2, 16); s2 += __shfl_xor(s2, 8); s2 += __shfl_xor(s2, 4);
  s2 += __shfl_xor(s2, 2);  s2 += __shfl_xor(s2, 1);
  if ((tid & 31) == 0) red[tid >> 5] = s2;
  __syncthreads();
  if (tid == 0) { float a = 0.f; for (int i = 0; i < 8; ++i) a += red[i]; bc[1] = a; }
  __syncthreads();

  const float rq  = rsqrtf(bc[0] * (1.f / QLORA_D)  + 1e-6f);
  const float rkv = rsqrtf(bc[1] * (1.f / KVLORA_D) + 1e-6f);
  for (int i = tid; i < QLORA_D; i += 256)
    qan[(size_t)t * QLORA_D + i] = (__bf16)(row[i] * rq * qw[i]);
  for (int i = tid; i < KVLORA_D; i += 256)
    kvan[(size_t)t * KVLORA_D + i] = (__bf16)(row[QLORA_D + i] * rkv * kvw[i]);

  if (tid < 32) {  // RoPE pairs of k_pe
    const float p = (float)pos[t];
    const float fr = p * __powf(10000.f, -(float)(2 * tid) * (1.f / 64.f));
    const float c = __cosf(fr), sn = __sinf(fr);
    const float x1 = row[QLORA_D + KVLORA_D + 2 * tid];
    const float x2 = row[QLORA_D + KVLORA_D + 2 * tid + 1];
    kpe[(size_t)t * ROPE_D + 2 * tid]     = x1 * c - x2 * sn;
    kpe[(size_t)t * ROPE_D + 2 * tid + 1] = x2 * c + x1 * sn;
  }
}

// ---------------------------------------------------------------------------
// q: RoPE the last 64 dims of each head, pre-scale by 192^-0.5, pack to bf16.
// ---------------------------------------------------------------------------
__global__ __launch_bounds__(256) void q_rope_pack_k(
    const float* __restrict__ q, const int* __restrict__ pos,
    __bf16* __restrict__ qfull) {
  const int t = blockIdx.x, tid = threadIdx.x;
  const float p = (float)pos[t];
  const float* qrow = q + (size_t)t * QB_N;
  for (int i = tid; i < QB_N; i += 256) {
    int d = i % QK_D;
    float v;
    if (d < NOPE_D) {
      v = qrow[i];
    } else {
      int base = i - d + NOPE_D;   // start of rope section for this head
      int pe = d - NOPE_D, pi = pe >> 1;
      float fr = p * __powf(10000.f, -(float)(2 * pi) * (1.f / 64.f));
      float c = __cosf(fr), sn = __sinf(fr);
      float x1 = qrow[base + 2 * pi], x2 = qrow[base + 2 * pi + 1];
      v = (pe & 1) ? (x2 * c + x1 * sn) : (x1 * c - x2 * sn);
    }
    qfull[(size_t)t * QB_N + i] = (__bf16)(v * ATTN_SCALE);
  }
}

// ---------------------------------------------------------------------------
// Build k_full (nope ++ broadcast roped k_pe) and v, packed bf16 (t,h,d).
// ---------------------------------------------------------------------------
__global__ __launch_bounds__(256) void kv_pack_k(
    const float* __restrict__ kv, const float* __restrict__ kpe,
    __bf16* __restrict__ kfull, __bf16* __restrict__ vbuf) {
  const int t = blockIdx.x, tid = threadIdx.x;
  for (int i = tid; i < NHEAD * QK_D; i += 256) {
    int hh = i / QK_D, d = i % QK_D;
    float v = (d < NOPE_D)
                  ? kv[(size_t)t * KVB_N + hh * (NOPE_D + VDIM_D) + d]
                  : kpe[(size_t)t * ROPE_D + (d - NOPE_D)];
    kfull[(size_t)t * (NHEAD * QK_D) + i] = (__bf16)v;
  }
  for (int i = tid; i < NHEAD * VDIM_D; i += 256) {
    int hh = i / VDIM_D, d = i % VDIM_D;
    vbuf[(size_t)t * (NHEAD * VDIM_D) + i] =
        (__bf16)kv[(size_t)t * KVB_N + hh * (NOPE_D + VDIM_D) + NOPE_D + d];
  }
}

// ---------------------------------------------------------------------------
// Causal flash attention. Grid (T/128, H), 256 threads = 8 waves.
// Wave w owns 16 query rows; key tiles of 64. Row softmax stats live in one
// half-wave -> 16-lane shfl_xor reductions only, no cross-wave traffic.
// K tile is staged by the Tensor Data Mover (iterate-mode D#: 64 rows of
// 192 bf16, global row stride 3072 elts, LDS row stride 200 elts).
// ---------------------------------------------------------------------------
__global__ __launch_bounds__(256) void mla_attn_k(
    const __bf16* __restrict__ qf, const __bf16* __restrict__ kf,
    const __bf16* __restrict__ vb, __bf16* __restrict__ attn) {
  __shared__ alignas(16) __bf16 lK[64][200];    // [key][feat 0..191]
  __shared__ alignas(16) __bf16 lVt[128][72];   // [dim][key]
  __shared__ alignas(16) __bf16 lP[8][16][72];  // per-wave P staging
  const int h = blockIdx.y;
  const int qb = blockIdx.x * 128;
  const int tid = threadIdx.x, wid = tid >> 5, lane = tid & 31;
  const int l16 = lane & 15, khalf = lane >> 4;
  const int qloc = wid * 16;

  // Q A-fragments straight from global (rows are 384B => 16B aligned chunks)
  v16bf qfrag[6];
  {
    const __bf16* qp = qf + ((size_t)(qb + qloc + l16) * NHEAD + h) * QK_D;
#pragma unroll
    for (int kc = 0; kc < 6; ++kc) qfrag[kc] = load_frag_a(qp + kc * 32, khalf);
  }

  float m_r[8], l_r[8];
  v8f Oa[8];
#pragma unroll
  for (int r = 0; r < 8; ++r) { m_r[r] = -1e30f; l_r[r] = 0.f; }
#pragma unroll
  for (int dt = 0; dt < 8; ++dt) Oa[dt] = v8f_zero();

  const int ktmax = (qb + 127) >> 6;
  for (int kt = 0; kt <= ktmax; ++kt) {
    const int sb = kt * 64;

#if defined(USE_TDM)
    if (wid == 0) {
      // --- Tensor DMA descriptor (D#) for the K tile -----------------------
      const unsigned long long gaddr =
          (unsigned long long)(const void*)(kf + ((size_t)sb * NHEAD + h) * QK_D);
      const unsigned lds_off =
          (unsigned)(unsigned long long)(const void*)&lK[0][0];
      v4u g0;
      g0[0] = 1u;                                   // count=1 (valid user D#)
      g0[1] = lds_off;                              // lds_addr (bytes)
      g0[2] = (unsigned)(gaddr & 0xffffffffu);      // global_addr[31:0]
      g0[3] = (unsigned)((gaddr >> 32) & 0x01ffffffu) | (2u << 30);  // +type=2
      v8i g1;
      g1[0] = (1 << 16) | (1 << 19);  // data_size=2B, iterate_enable=1
      g1[1] = (QK_D << 16);           // tensor_dim0[15:0]=192 (bits 63:48)
      g1[2] = (64 << 16);             // tensor_dim1[15:0]=64  (bits 95:80)
      g1[3] = (QK_D << 16);           // tile_dim0=192         (bits 127:112)
      g1[4] = 1;                      // tile_dim1=1, tile_dim2=0
      g1[5] = NHEAD * QK_D;           // tensor_dim0_stride[31:0]=3072
      g1[6] = 0;                      // stride0[47:32]=0, dim1_stride lo=0
      g1[7] = 0;
      v4i g2;
      g2[0] = 1;                      // tensor_dim2
      g2[1] = 200;                    // lds_addr_increment (elements)
      g2[2] = NHEAD * QK_D;           // global_addr_increment (elements)
      g2[3] = (63 << 16);             // iterate_count=63 (64 rows)
      v4i g3;
      g3[0] = 0; g3[1] = 0; g3[2] = 0; g3[3] = 0;
#if defined(__clang_major__) && (__clang_major__ >= 23)
      v8i g4;
#pragma unroll
      for (int i = 0; i < 8; ++i) g4[i] = 0;
      __builtin_amdgcn_tensor_load_to_lds(g0, g1, g2, g3, g4, 0);
#else
      __builtin_amdgcn_tensor_load_to_lds(g0, g1, g2, g3, 0);
#endif
    }
#else
    // Fallback: stage K tile 64 x 192 with vector loads (dword pairs)
    for (int i = tid; i < 64 * 96; i += 256) {
      int s = i / 96, cp = i % 96;
      v2bf v = *(const v2bf*)(kf + ((size_t)(sb + s) * NHEAD + h) * QK_D + cp * 2);
      *(v2bf*)&lK[s][cp * 2] = v;
    }
#endif

    // stage V tile 64 x 128 transposed into [dim][key] (all waves, overlaps TDM)
    for (int i = tid; i < 64 * 64; i += 256) {
      int s = i / 64, dp = i % 64;
      v2bf v = *(const v2bf*)(vb + ((size_t)(sb + s) * NHEAD + h) * VDIM_D + dp * 2);
      lVt[dp * 2][s]     = v[0];
      lVt[dp * 2 + 1][s] = v[1];
    }

#if defined(USE_TDM)
    if (wid == 0) __builtin_amdgcn_s_wait_tensorcnt(0);
#endif
    __syncthreads();

    // scores: Q(16x192) @ K^T -> 16x64 in 4 accumulators
    v8f sc[4];
#pragma unroll
    for (int nt = 0; nt < 4; ++nt) sc[nt] = v8f_zero();
#pragma unroll
    for (int nt = 0; nt < 4; ++nt)
#pragma unroll
      for (int kc = 0; kc < 6; ++kc) {
        v16bf b = load_frag_b(&lK[nt * 16 + l16][kc * 32], khalf);
        sc[nt] = wmma_bf16(qfrag[kc], b, sc[nt]);
      }

    // causal mask + online softmax (per row, 16-lane reductions)
#pragma unroll
    for (int r = 0; r < 8; ++r) {
      const int row = qb + qloc + khalf * 8 + r;
      float pmax = -1e30f;
#pragma unroll
      for (int nt = 0; nt < 4; ++nt) {
        const int col = sb + nt * 16 + l16;
        float v = sc[nt][r];
        v = (col > row) ? -1e30f : v;
        sc[nt][r] = v;
        pmax = fmaxf(pmax, v);
      }
      pmax = fmaxf(pmax, __shfl_xor(pmax, 1));
      pmax = fmaxf(pmax, __shfl_xor(pmax, 2));
      pmax = fmaxf(pmax, __shfl_xor(pmax, 4));
      pmax = fmaxf(pmax, __shfl_xor(pmax, 8));
      const float mn = fmaxf(m_r[r], pmax);
      const float corr = __expf(m_r[r] - mn);
      m_r[r] = mn;
      float psum = 0.f;
#pragma unroll
      for (int nt = 0; nt < 4; ++nt) {
        float pv = __expf(sc[nt][r] - mn);
        sc[nt][r] = pv;
        psum += pv;
      }
      psum += __shfl_xor(psum, 1);
      psum += __shfl_xor(psum, 2);
      psum += __shfl_xor(psum, 4);
      psum += __shfl_xor(psum, 8);
      l_r[r] = l_r[r] * corr + psum;
#pragma unroll
      for (int dt = 0; dt < 8; ++dt) Oa[dt][r] *= corr;
    }

    // P accumulator-layout -> A-fragment layout via per-wave LDS
#pragma unroll
    for (int nt = 0; nt < 4; ++nt)
#pragma unroll
      for (int r = 0; r < 8; ++r)
        lP[wid][khalf * 8 + r][nt * 16 + l16] = (__bf16)sc[nt][r];
    __syncthreads();

    v16bf pf[2];
#pragma unroll
    for (int kc = 0; kc < 2; ++kc)
      pf[kc] = load_frag_a(&lP[wid][l16][kc * 32], khalf);
#pragma unroll
    for (int dt = 0; dt < 8; ++dt)
#pragma unroll
      for (int kc = 0; kc < 2; ++kc) {
        v16bf b = load_frag_b(&lVt[dt * 16 + l16][kc * 32], khalf);
        Oa[dt] = wmma_bf16(pf[kc], b, Oa[dt]);
      }
    __syncthreads();  // all reads of lK/lVt/lP done before next stage
  }

  // normalize and emit attn (t,h,d) bf16
#pragma unroll
  for (int r = 0; r < 8; ++r) {
    const int row = qb + qloc + khalf * 8 + r;
    const float inv = 1.f / l_r[r];
#pragma unroll
    for (int dt = 0; dt < 8; ++dt)
      attn[((size_t)row * NHEAD + h) * VDIM_D + dt * 16 + l16] =
          (__bf16)(Oa[dt][r] * inv);
  }
}

// ---------------------------------------------------------------------------
extern "C" void kernel_launch(void* const* d_in, const int* in_sizes, int n_in,
                              void* d_out, int out_size, void* d_ws, size_t ws_size,
                              hipStream_t stream) {
  (void)in_sizes; (void)n_in; (void)out_size; (void)ws_size;
  const int*   positions = (const int*)d_in[0];
  const float* hidden    = (const float*)d_in[1];
  const float* w_fused   = (const float*)d_in[2];
  const float* q_a_ln_w  = (const float*)d_in[3];
  const float* kv_a_ln_w = (const float*)d_in[4];
  const float* w_qb      = (const float*)d_in[5];
  const float* w_kvb     = (const float*)d_in[6];
  const float* w_o       = (const float*)d_in[7];
  float* out = (float*)d_out;

  char* wptr = (char*)d_ws;
  auto take = [&](size_t bytes) -> char* {
    char* p = wptr;
    wptr += (bytes + 255) & ~(size_t)255;
    return p;
  };
  float*  qkv_a = (float*) take((size_t)T_LEN * FUSED_N * 4);
  float*  qmat  = (float*) take((size_t)T_LEN * QB_N   * 4);
  float*  kvmat = (float*) take((size_t)T_LEN * KVB_N  * 4);
  __bf16* qan   = (__bf16*)take((size_t)T_LEN * QLORA_D * 2);
  __bf16* kvan  = (__bf16*)take((size_t)T_LEN * KVLORA_D * 2);
  float*  kpe   = (float*) take((size_t)T_LEN * ROPE_D * 4);
  __bf16* qfull = (__bf16*)take((size_t)T_LEN * QB_N * 2);
  __bf16* kfull = (__bf16*)take((size_t)T_LEN * NHEAD * QK_D * 2);
  __bf16* vbuf  = (__bf16*)take((size_t)T_LEN * NHEAD * VDIM_D * 2);
  __bf16* attn  = (__bf16*)take((size_t)T_LEN * NHEAD * VDIM_D * 2);

  dim3 blk(256);
  // 1) qkv_a = hidden @ w_fused          (2048 x 2112, K=5120)
  gemm_wb<float><<<dim3((FUSED_N + 127) / 128, T_LEN / 128), blk, 0, stream>>>(
      hidden, w_fused, qkv_a, T_LEN, FUSED_N, HID_D);
  // 2) rmsnorm q_a/kv_a + rope k_pe
  rms_rope_k<<<dim3(T_LEN), blk, 0, stream>>>(qkv_a, q_a_ln_w, kv_a_ln_w,
                                              positions, qan, kvan, kpe);
  // 3) q = q_a_n @ w_qb                  (2048 x 3072, K=1536)
  gemm_wb<__bf16><<<dim3(QB_N / 128, T_LEN / 128), blk, 0, stream>>>(
      qan, w_qb, qmat, T_LEN, QB_N, QLORA_D);
  // 4) kv = kv_a_n @ w_kvb               (2048 x 4096, K=512)
  gemm_wb<__bf16><<<dim3(KVB_N / 128, T_LEN / 128), blk, 0, stream>>>(
      kvan, w_kvb, kvmat, T_LEN, KVB_N, KVLORA_D);
  // 5) rope + scale + pack q_full
  q_rope_pack_k<<<dim3(T_LEN), blk, 0, stream>>>(qmat, positions, qfull);
  // 6) pack k_full / v
  kv_pack_k<<<dim3(T_LEN), blk, 0, stream>>>(kvmat, kpe, kfull, vbuf);
  // 7) causal flash attention (TDM-staged K tiles)
  mla_attn_k<<<dim3(T_LEN / 128, NHEAD), blk, 0, stream>>>(qfull, kfull, vbuf, attn);
  // 8) out = attn @ w_o                  (2048 x 5120, K=2048)
  gemm_wb<__bf16><<<dim3(HID_D / 128, T_LEN / 128), blk, 0, stream>>>(
      attn, w_o, out, T_LEN, HID_D, NHEAD * VDIM_D);
}
